// ModelVT_32143535243332
// MI455X (gfx1250) — matrix operand order repaired and checked by reading, not compiled, not run on hardware
//
#include <hip/hip_runtime.h>

typedef unsigned short u16;
typedef __attribute__((ext_vector_type(16))) __bf16 v16bf;
typedef __attribute__((ext_vector_type(8)))  float  v8f;

namespace {

constexpr int kB  = 4096, kH = 128, kW = 128;
constexpr int kGS = 16, kNG = 6, kNS = 3, kLS = 2, kNC = 10;
constexpr int kFS = 128, kGFS = 256, kHS = 256;
constexpr int kINP = kGS * kGS * kNS;            // 768
constexpr float kSTD = 0.1f;
// output offsets (floats): act_log_probs, locs, logps, cvals
constexpr int kOAct = 0;
constexpr int kOLoc = kB * kNC;                  // 40960
constexpr int kOLogp = kOLoc + kB * kNG * kLS;   // 90112
constexpr int kOCval = kOLogp + kB * kNG;        // 114688

__device__ __forceinline__ u16 f2bf(float f) {
  unsigned u = __float_as_uint(f);
  u += 0x7fffu + ((u >> 16) & 1u);               // round-to-nearest-even
  return (u16)(u >> 16);
}
__device__ __forceinline__ float sigmoidf(float x) { return 1.f / (1.f + expf(-x)); }

// ---------------------------------------------------------------------------
// Weight swizzle: fp32 (K x N, optionally two stacked sources split at Ksplit)
// -> bf16 in WMMA B-fragment order. Tile (kt,nt) covers K=kt*32..+31,
// N=nt*16..+15. Within a tile: lane<16 holds K=0..15 of column N=lane,
// lane>=16 holds K=16..31 of column N=lane-16; 16 bf16 contiguous per lane.
// ---------------------------------------------------------------------------
__global__ void swizzle_w(const float* __restrict__ W0, const float* __restrict__ W1,
                          int Ksplit, int N, u16* __restrict__ out) {
  int idx = blockIdx.x * blockDim.x + threadIdx.x;   // K*N total
  int i    = idx & 15;
  int lane = (idx >> 4) & 31;
  int tile = idx >> 9;
  int ntN  = N >> 4;
  int nt   = tile % ntN;
  int kt   = tile / ntN;
  int k = kt * 32 + (lane >> 4) * 16 + i;
  int n = nt * 16 + (lane & 15);
  float w = (k < Ksplit) ? W0[(size_t)k * N + n] : W1[(size_t)(k - Ksplit) * N + n];
  out[idx] = f2bf(w);
}

__global__ void bias_prep(const float* b_gg, const float* b_lg,
                          const float* b_ih, const float* b_hh,
                          float* bcat, float* bcat2) {
  int idx = blockIdx.x * blockDim.x + threadIdx.x;   // 1024
  bcat2[idx] = b_ih[idx] + b_hh[idx];
  if (idx < kGFS) bcat[idx] = b_gg[idx] + b_lg[idx];
}

__global__ void init_state(float* c, u16* u2, float* loc, const float* loc_init) {
  int idx = blockIdx.x * blockDim.x + threadIdx.x;   // kB*kHS
  c[idx] = 0.f;
  int b = idx >> 8, j = idx & 255;
  u2[(size_t)b * 512 + 256 + j] = 0;                 // h0 = 0 (bf16 half of [gt;h])
  if (idx < kB * kLS) loc[idx] = loc_init[idx];
}

// ---------------------------------------------------------------------------
// Retina: bilinear-sample 3 scales of 16x16 glimpses around loc -> gl bf16
// column layout: c = s*256 + y*16 + x  (matches reference reshape/concat)
// ---------------------------------------------------------------------------
__global__ void retina(const float* __restrict__ img, const float* __restrict__ loc,
                       u16* __restrict__ gl) {
  int idx = blockIdx.x * blockDim.x + threadIdx.x;   // kB * kINP
  int b = idx / kINP;
  int c = idx - b * kINP;
  int s  = c >> 8;
  int r  = c & 255;
  int yi = r >> 4, xi = r & 15;
  float base = (float)kGS / (float)(kH - 1);
  float scale = base * (float)(1 << s);
  float xs = -1.f + 2.f * (float)xi / 15.f;
  float ys = -1.f + 2.f * (float)yi / 15.f;
  float gx = scale * xs + loc[b * 2 + 0];
  float gy = scale * ys + loc[b * 2 + 1];
  float x = (gx + 1.f) * 0.5f * (float)(kW - 1);
  float y = (gy + 1.f) * 0.5f * (float)(kH - 1);
  float xf = floorf(x), yf = floorf(y);
  int x0 = (int)xf, y0 = (int)yf;
  float wx = x - xf, wy = y - yf;
  const float* ib = img + (size_t)b * kH * kW;
  auto tap = [&](int iy, int ix) -> float {
    if (ix < 0 || ix >= kW || iy < 0 || iy >= kH) return 0.f;
    return ib[iy * kW + ix];
  };
  float v = tap(y0, x0)     * (1.f - wx) * (1.f - wy)
          + tap(y0, x0 + 1) * wx         * (1.f - wy)
          + tap(y0 + 1, x0) * (1.f - wx) * wy
          + tap(y0 + 1, x0 + 1) * wx     * wy;
  gl[(size_t)b * kINP + c] = f2bf(v);
}

// hl = relu(loc @ w_l + b_l) -> bf16 into columns [128,256) of u1 (ld 256)
__global__ void loc_feat(const float* __restrict__ loc, const float* __restrict__ w_l,
                         const float* __restrict__ b_l, u16* __restrict__ u1) {
  int idx = blockIdx.x * blockDim.x + threadIdx.x;   // kB * kFS
  int b = idx >> 7, j = idx & 127;
  float v = loc[b * 2 + 0] * w_l[j] + loc[b * 2 + 1] * w_l[kFS + j] + b_l[j];
  u1[(size_t)b * 256 + 128 + j] = f2bf(fmaxf(v, 0.f));
}

// ---------------------------------------------------------------------------
// bf16 WMMA GEMM: C[M,N] = A[M,K] * Bsw + bias, M=4096 fixed by grid.
// Block = 8 waves; wave computes 16x64 tile (4 x v_wmma_f32_16x16x32_bf16
// per K-step of 32). A fragment in contiguous k order (lane half hi: k=16*hi..+15, same as B); B fragments are
// straight 32B contiguous loads from the pre-swizzled weights (L2-resident).
// ---------------------------------------------------------------------------
__global__ __launch_bounds__(256) void gemm_bf16_wmma(
    const u16* __restrict__ A, int lda,
    const u16* __restrict__ Bsw, int N, int K,
    const float* __restrict__ bias,
    float* __restrict__ outF, int ldf,
    u16* __restrict__ outB, int ldb, int doRelu) {
  const int lane = threadIdx.x & 31;
  const int wave = threadIdx.x >> 5;
  const int m0 = blockIdx.x * 128 + wave * 16;
  const int n0 = blockIdx.y * 64;
  const int l15 = lane & 15;
  const int hi  = lane >> 4;
  const int ntN = N >> 4;

  v8f acc[4] = {};

  const u16* arow = A + (size_t)(m0 + l15) * lda + hi * 16;
  const u16* bcol = Bsw + ((size_t)(n0 >> 4) * 32 + lane) * 16;
  const size_t bstepK = (size_t)ntN * 512;           // elements per kt step

  for (int k0 = 0; k0 < K; k0 += 32) {
    union { v16bf v; uint4 q[2]; } a;
    a.q[0] = *(const uint4*)(arow + k0);             // K: k0 + hi*16 + 0..7
    a.q[1] = *(const uint4*)(arow + k0 + 8);         // K: k0 + hi*16 + 8..15
#pragma unroll
    for (int j = 0; j < 4; ++j) {
      union { v16bf v; uint4 q[2]; } bf;
      const u16* bp = bcol + (size_t)j * 512;
      bf.q[0] = *(const uint4*)(bp);
      bf.q[1] = *(const uint4*)(bp + 8);
      acc[j] = __builtin_amdgcn_wmma_f32_16x16x32_bf16(
          false, a.v, false, bf.v, (short)0, acc[j], false, false);
    }
    bcol += bstepK;
  }

  const int rowBase = m0 + hi * 8;                   // C layout: VGPR r -> M=r / M=8+r
#pragma unroll
  for (int j = 0; j < 4; ++j) {
    const int col = n0 + j * 16 + l15;
    const float bs = bias ? bias[col] : 0.f;
#pragma unroll
    for (int r = 0; r < 8; ++r) {
      float v = acc[j][r] + bs;
      if (doRelu) v = fmaxf(v, 0.f);
      const int row = rowBase + r;
      if (outF) outF[(size_t)row * ldf + col] = v;
      if (outB) outB[(size_t)row * ldb + col] = f2bf(v);
    }
  }
}

// LSTM cell (gate order i,f,g,o); writes h fp32 + h bf16 into u2[:,256:512]
__global__ void lstm_cell(const float* __restrict__ gates, float* __restrict__ c,
                          float* __restrict__ h, u16* __restrict__ u2) {
  int idx = blockIdx.x * blockDim.x + threadIdx.x;   // kB * kHS
  int b = idx >> 8, j = idx & 255;
  const float* g = gates + (size_t)b * 1024;
  float i_ = g[j], f_ = g[256 + j], g_ = g[512 + j], o_ = g[768 + j];
  float cc = sigmoidf(f_) * c[idx] + sigmoidf(i_) * tanhf(g_);
  float hh = sigmoidf(o_) * tanhf(cc);
  c[idx] = cc;
  h[idx] = hh;
  u2[(size_t)b * 512 + 256 + j] = f2bf(hh);
}

// Per-row heads: mu = tanh(h@w_loc+b_loc), sample/logp/clip, cval = h@w_crit
__global__ __launch_bounds__(256) void loc_head(
    const float* __restrict__ h, const float* __restrict__ w_loc,
    const float* __restrict__ b_loc, const float* __restrict__ w_crit,
    const float* __restrict__ b_crit, const float* __restrict__ noise,
    int t, float* __restrict__ loc, float* __restrict__ out) {
  __shared__ float s0[256], s1[256], s2[256];
  int b = blockIdx.x, tid = threadIdx.x;
  float hv = h[(size_t)b * kHS + tid];
  s0[tid] = hv * w_loc[tid * 2 + 0];
  s1[tid] = hv * w_loc[tid * 2 + 1];
  s2[tid] = hv * w_crit[tid];
  __syncthreads();
  for (int s = 128; s > 0; s >>= 1) {
    if (tid < s) { s0[tid] += s0[tid + s]; s1[tid] += s1[tid + s]; s2[tid] += s2[tid + s]; }
    __syncthreads();
  }
  if (tid == 0) {
    float mux = tanhf(s0[0] + b_loc[0]);
    float muy = tanhf(s1[0] + b_loc[1]);
    float nx = noise[((size_t)t * kB + b) * 2 + 0];
    float ny = noise[((size_t)t * kB + b) * 2 + 1];
    float sx = mux + kSTD * nx, sy = muy + kSTD * ny;
    // logp = sum_d(-0.5*noise^2 - log(STD) - 0.5*log(2pi))
    float logp = -0.5f * (nx * nx + ny * ny) + 2.7672931195787458f;
    float lx = fminf(fmaxf(sx, -1.f), 1.f);
    float ly = fminf(fmaxf(sy, -1.f), 1.f);
    loc[b * 2 + 0] = lx;
    loc[b * 2 + 1] = ly;
    out[kOLoc + ((size_t)b * kNG + t) * 2 + 0] = lx;
    out[kOLoc + ((size_t)b * kNG + t) * 2 + 1] = ly;
    out[kOLogp + (size_t)b * kNG + t] = logp;
    out[kOCval + (size_t)b * kNG + t] = s2[0] + b_crit[0];
  }
}

// act_log_probs = log_softmax(h @ w_act + b_act)
__global__ __launch_bounds__(256) void classifier(
    const float* __restrict__ h, const float* __restrict__ w_act,
    const float* __restrict__ b_act, float* __restrict__ out) {
  __shared__ float red[256];
  __shared__ float logits[kNC];
  __shared__ float lse;
  int b = blockIdx.x, tid = threadIdx.x;
  float hv = h[(size_t)b * kHS + tid];
  for (int k = 0; k < kNC; ++k) {
    red[tid] = hv * w_act[tid * kNC + k];
    __syncthreads();
    for (int s = 128; s > 0; s >>= 1) {
      if (tid < s) red[tid] += red[tid + s];
      __syncthreads();
    }
    if (tid == 0) logits[k] = red[0] + b_act[k];
    __syncthreads();
  }
  if (tid == 0) {
    float mx = logits[0];
    for (int k = 1; k < kNC; ++k) mx = fmaxf(mx, logits[k]);
    float se = 0.f;
    for (int k = 0; k < kNC; ++k) se += expf(logits[k] - mx);
    lse = mx + logf(se);
  }
  __syncthreads();
  if (tid < kNC) out[kOAct + (size_t)b * kNC + tid] = logits[tid] - lse;
}

}  // namespace

extern "C" void kernel_launch(void* const* d_in, const int* in_sizes, int n_in,
                              void* d_out, int out_size, void* d_ws, size_t ws_size,
                              hipStream_t stream) {
  (void)in_sizes; (void)n_in; (void)out_size; (void)ws_size;
  const float* data     = (const float*)d_in[0];
  const float* loc_init = (const float*)d_in[1];
  const float* noise    = (const float*)d_in[2];
  const float* w_g  = (const float*)d_in[3];
  const float* b_g  = (const float*)d_in[4];
  const float* w_l  = (const float*)d_in[5];
  const float* b_l  = (const float*)d_in[6];
  const float* w_gg = (const float*)d_in[7];
  const float* b_gg = (const float*)d_in[8];
  const float* w_lg = (const float*)d_in[9];
  const float* b_lg = (const float*)d_in[10];
  const float* w_ih = (const float*)d_in[11];
  const float* b_ih = (const float*)d_in[12];
  const float* w_hh = (const float*)d_in[13];
  const float* b_hh = (const float*)d_in[14];
  const float* w_loc  = (const float*)d_in[15];
  const float* b_loc  = (const float*)d_in[16];
  const float* w_act  = (const float*)d_in[17];
  const float* b_act  = (const float*)d_in[18];
  const float* w_crit = (const float*)d_in[19];
  const float* b_crit = (const float*)d_in[20];
  float* out = (float*)d_out;

  char* p = (char*)d_ws;
  auto alloc = [&](size_t bytes) { void* r = (void*)p; p += (bytes + 255) & ~(size_t)255; return r; };
  u16*   gl    = (u16*)  alloc((size_t)kB * kINP * 2);   // glimpse, bf16
  u16*   u1    = (u16*)  alloc((size_t)kB * 256 * 2);    // [hg ; hl] bf16
  u16*   u2    = (u16*)  alloc((size_t)kB * 512 * 2);    // [gt ; h]  bf16
  float* gates = (float*)alloc((size_t)kB * 1024 * 4);
  float* hbuf  = (float*)alloc((size_t)kB * kHS * 4);
  float* cbuf  = (float*)alloc((size_t)kB * kHS * 4);
  float* locb  = (float*)alloc((size_t)kB * kLS * 4);
  u16*   wg_sw    = (u16*)alloc((size_t)kINP * kFS * 2);
  u16*   wcat_sw  = (u16*)alloc((size_t)256 * kGFS * 2); // [w_gg; w_lg]
  u16*   wcat2_sw = (u16*)alloc((size_t)512 * 1024 * 2); // [w_ih; w_hh]
  float* bcat  = (float*)alloc(kGFS * 4);
  float* bcat2 = (float*)alloc(1024 * 4);

  // One-time (per launch) weight prep: bf16 + WMMA B-fragment swizzle
  swizzle_w<<<(kINP * kFS) / 256, 256, 0, stream>>>(w_g, w_g, kINP, kFS, wg_sw);
  swizzle_w<<<(256 * kGFS) / 256, 256, 0, stream>>>(w_gg, w_lg, kFS, kGFS, wcat_sw);
  swizzle_w<<<(512 * 1024) / 256, 256, 0, stream>>>(w_ih, w_hh, kGFS, 1024, wcat2_sw);
  bias_prep<<<4, 256, 0, stream>>>(b_gg, b_lg, b_ih, b_hh, bcat, bcat2);
  init_state<<<(kB * kHS) / 256, 256, 0, stream>>>(cbuf, u2, locb, loc_init);

  for (int t = 0; t < kNG; ++t) {
    retina<<<(kB * kINP) / 256, 256, 0, stream>>>(data, locb, gl);
    loc_feat<<<(kB * kFS) / 256, 256, 0, stream>>>(locb, w_l, b_l, u1);
    // hg = relu(gl @ w_g + b_g)        -> u1[:, 0:128)   (bf16)
    gemm_bf16_wmma<<<dim3(kB / 128, kFS / 64), 256, 0, stream>>>(
        gl, kINP, wg_sw, kFS, kINP, b_g, nullptr, 0, u1, 256, 1);
    // gt = relu([hg;hl] @ [w_gg;w_lg] + bcat) -> u2[:, 0:256) (bf16)
    gemm_bf16_wmma<<<dim3(kB / 128, kGFS / 64), 256, 0, stream>>>(
        u1, 256, wcat_sw, kGFS, 256, bcat, nullptr, 0, u2, 512, 1);
    // gates = [gt;h] @ [w_ih;w_hh] + bcat2 -> fp32
    gemm_bf16_wmma<<<dim3(kB / 128, 1024 / 64), 256, 0, stream>>>(
        u2, 512, wcat2_sw, 1024, 512, bcat2, gates, 1024, nullptr, 0, 0);
    lstm_cell<<<(kB * kHS) / 256, 256, 0, stream>>>(gates, cbuf, hbuf, u2);
    loc_head<<<kB, 256, 0, stream>>>(hbuf, w_loc, b_loc, w_crit, b_crit, noise, t, locb, out);
  }
  classifier<<<kB, 256, 0, stream>>>(hbuf, w_act, b_act, out);
}
